// BinaryDecoderV2_42734924595812
// MI455X (gfx1250) — matrix operand order repaired and checked by reading, not compiled, not run on hardware
//
#include <hip/hip_runtime.h>
#include <hip/hip_bf16.h>
#include <hip/hip_fp16.h>

// ---------------------------------------------------------------------------
// Problem constants (from reference)
// ---------------------------------------------------------------------------
#define M_DIM 2048   // batch
#define K_DIM 8192   // IN_FEATURES
#define N_DIM 1024   // OUT_FEATURES

typedef __attribute__((ext_vector_type(16))) _Float16 v16h;
typedef __attribute__((ext_vector_type(8)))  float    v8f;
typedef __attribute__((ext_vector_type(4)))  int      v4i;

#define AS1 __attribute__((address_space(1)))
#define AS3 __attribute__((address_space(3)))

// POD 16-byte vector (HIP's uint4 has ctors -> illegal in unions)
struct alignas(16) U4 { unsigned x, y, z, w; };
struct alignas(8)  H4 { _Float16 a, b, c, d; };

// ---------------------------------------------------------------------------
// CDNA5 async global->LDS staging (ASYNCcnt-tracked, bypasses VGPRs).
// Builtin signature (from hipcc diagnostic): param0 = v4i __device__*,
// i.e. typed 128-bit element pointers.
// ---------------------------------------------------------------------------
#if defined(__has_builtin)
#  if __has_builtin(__builtin_amdgcn_global_load_async_to_lds_b128)
#    define HAVE_ASYNC_LDS 1
#  endif
#endif
#ifndef HAVE_ASYNC_LDS
#  define HAVE_ASYNC_LDS 0
#endif

__device__ __forceinline__ void stage_b128(const _Float16* g, _Float16* l) {
#if HAVE_ASYNC_LDS
    __builtin_amdgcn_global_load_async_to_lds_b128(
        (AS1 v4i*)g, (AS3 v4i*)l, 0, 0);
#else
    *(U4*)l = *(const U4*)g;
#endif
}

__device__ __forceinline__ void stage_fence() {
#if HAVE_ASYNC_LDS
#  if __has_builtin(__builtin_amdgcn_s_wait_asynccnt)
    __builtin_amdgcn_s_wait_asynccnt(0);
#  else
    asm volatile("s_wait_asynccnt 0x0" ::: "memory");
#  endif
#endif
}

__device__ __forceinline__ _Float16 pack8(const float4 lo, const float4 hi) {
    int u = 0;
    u |= (lo.x > 0.f) ?   1 : 0;
    u |= (lo.y > 0.f) ?   2 : 0;
    u |= (lo.z > 0.f) ?   4 : 0;
    u |= (lo.w > 0.f) ?   8 : 0;
    u |= (hi.x > 0.f) ?  16 : 0;
    u |= (hi.y > 0.f) ?  32 : 0;
    u |= (hi.z > 0.f) ?  64 : 0;
    u |= (hi.w > 0.f) ? 128 : 0;
    return (_Float16)(float)(int)(signed char)u;   // exact in f16 (|v| <= 128)
}

// ---------------------------------------------------------------------------
// Kernel: pack weight [K, N*8] f32 -> B COLUMN-MAJOR [N, K] f16 holding exact
// signed-int8 values (bit b = (w > 0), powers = [1,2,...,64,-128]).
// Transpose done once here (through LDS) so the GEMM needs none.
// ---------------------------------------------------------------------------
__global__ void __launch_bounds__(256)
pack_weight_kernel(const float* __restrict__ w, _Float16* __restrict__ bp) {
    __shared__ alignas(16) _Float16 T[64 * 72];   // [n_local][k_local], padded

    const int tid = threadIdx.x;
    const int k0  = blockIdx.x * 64;              // K_DIM/64 = 128 tiles
    const int n0  = blockIdx.y * 64;              // N_DIM/64 = 16 tiles

    // read phase: thread -> (k_local = tid>>2, n quarter = tid&3)
    const int kl = tid >> 2;
    const int q  = tid & 3;
    const float* wrow = w + (size_t)(k0 + kl) * (N_DIM * 8) + (size_t)n0 * 8;
#pragma unroll
    for (int j = 0; j < 16; ++j) {
        const float4* w4 = (const float4*)(wrow + (q * 16 + j) * 8);
        T[(q * 16 + j) * 72 + kl] = pack8(w4[0], w4[1]);
    }
    __syncthreads();

    // write phase: thread -> (n_local = tid>>2, k chunk = (tid&3)*16)
    const int nl = tid >> 2;
    const int kq = (tid & 3) * 16;
    const U4* src = (const U4*)&T[nl * 72 + kq];
    U4* dst = (U4*)(bp + (size_t)(n0 + nl) * K_DIM + k0 + kq);
    dst[0] = src[0];
    dst[1] = src[1];
}

// ---------------------------------------------------------------------------
// Kernel: latent f32 -> f16 (A matrix), 4 elements / thread
// ---------------------------------------------------------------------------
__global__ void cvt_latent_kernel(const float* __restrict__ x,
                                  _Float16* __restrict__ y) {
    const size_t i = (size_t)(blockIdx.x * blockDim.x + threadIdx.x) * 4;
    const float4 v = *(const float4*)(x + i);
    H4 h = { (_Float16)v.x, (_Float16)v.y, (_Float16)v.z, (_Float16)v.w };
    *(H4*)(y + i) = h;
}

// ---------------------------------------------------------------------------
// Kernel: true_sum [M, N*8] f32 -> int_sum [M, N] f32 (dot with powers)
// ---------------------------------------------------------------------------
__global__ void int_sum_kernel(const float* __restrict__ t,
                               float* __restrict__ s) {
    const int idx = blockIdx.x * blockDim.x + threadIdx.x;   // m*N + n
    const float4* t4 = (const float4*)(t + (size_t)idx * 8);
    const float4 lo = t4[0];
    const float4 hi = t4[1];
    s[idx] = lo.x + 2.f * lo.y + 4.f * lo.z + 8.f * lo.w
           + 16.f * hi.x + 32.f * hi.y + 64.f * hi.z - 128.f * hi.w;
}

// ---------------------------------------------------------------------------
// GEMM + fused loss: pred = A(f16) @ B(f16), f32 WMMA accumulation,
// loss partial = sum((pred - int_sum)^2), atomically accumulated.
//
// Block tile 128x128, BK=64, 256 threads = 8 wave32 (4M x 2N wave grid),
// 2x4 16x16 accumulators per wave -> 16 WMMAs / wave / iteration.
// LDS double-buffered; next tile staged via async global->LDS DMA while the
// current tile is consumed; one s_wait_asynccnt + barrier per iteration.
// B is pre-transposed ([N,K]) so A and B staging are identical b128 copies.
// ---------------------------------------------------------------------------
#define BM  128
#define BN  128
#define BK  64
#define LDS 72   // padded stride in halves (144B: conflict-free 16B reads)

__global__ void __launch_bounds__(256)
gemm_loss_kernel(const _Float16* __restrict__ A,
                 const _Float16* __restrict__ B,
                 const float* __restrict__ S,
                 float* __restrict__ loss_acc) {
    __shared__ alignas(16) _Float16 As0[BM * LDS];
    __shared__ alignas(16) _Float16 As1[BM * LDS];
    __shared__ alignas(16) _Float16 Bs0[BN * LDS];
    __shared__ alignas(16) _Float16 Bs1[BN * LDS];
    __shared__ float red[8];

    const int tid  = threadIdx.x;
    const int lane = tid & 31;
    const int wave = tid >> 5;
    const int wm   = wave & 3;    // 32 M-rows per wave group
    const int wn   = wave >> 2;   // 64 N-cols per wave group
    const int bm0  = blockIdx.y * BM;
    const int bn0  = blockIdx.x * BN;

    // staging: each thread copies 64B of A and 64B of B per K-step
    const int row  = tid >> 1;            // 0..127 (A m-row / B n-row)
    const int coff = (tid & 1) * 32;      // half offset within the 64-half row
    const int sofs = row * LDS + coff;    // this thread's LDS staging slot

    const _Float16* Ag = A + (size_t)(bm0 + row) * K_DIM + coff;
    const _Float16* Bg = B + (size_t)(bn0 + row) * K_DIM + coff;

    const v8f vzero = { 0.f, 0.f, 0.f, 0.f, 0.f, 0.f, 0.f, 0.f };
    v8f acc[2][4];
#pragma unroll
    for (int i = 0; i < 2; ++i)
#pragma unroll
        for (int j = 0; j < 4; ++j) acc[i][j] = vzero;

    // WMMA f16 operand lane layout (wave32):
    //  A (16x32): lanes 0-15 -> K {0..7,16..23}; lanes 16-31 -> K {8..15,24..31}
    //  B (32x16): lane%16 = N; lanes 0-15 -> K {0..7,16..23}; lanes 16-31 -> K {8..15,24..31}
    const int kbA   = (lane < 16) ? 0 : 8;
    const int kbB   = (lane < 16) ? 0 : 8;
    const int arow0 = (wm * 32 + (lane & 15)) * LDS;
    const int bcol0 = (wn * 64 + (lane & 15)) * LDS;

    union Frag { U4 u[2]; v16h h; };

    _Float16* curA = As0; _Float16* nxtA = As1;
    _Float16* curB = Bs0; _Float16* nxtB = Bs1;

    // prologue: stage tile 0 into the current buffers
#pragma unroll
    for (int j = 0; j < 4; ++j) {
        stage_b128(Ag + j * 8, curA + sofs + j * 8);
        stage_b128(Bg + j * 8, curB + sofs + j * 8);
    }
    stage_fence();
    __syncthreads();

    for (int k0 = 0; k0 < K_DIM; k0 += BK) {
        const int kn = k0 + BK;
        if (kn < K_DIM) {
            // DMA next tile into the other buffers; overlaps the WMMAs below
#pragma unroll
            for (int j = 0; j < 4; ++j) {
                stage_b128(Ag + kn + j * 8, nxtA + sofs + j * 8);
                stage_b128(Bg + kn + j * 8, nxtB + sofs + j * 8);
            }
        }

        // compute: 2 K-substeps of 32 from the current buffers
#pragma unroll
        for (int ks = 0; ks < 2; ++ks) {
            const int kof = ks * 32;
            Frag af[2], bf[4];
#pragma unroll
            for (int tm = 0; tm < 2; ++tm) {
                const _Float16* p = curA + arow0 + tm * 16 * LDS + kof + kbA;
                af[tm].u[0] = *(const U4*)(p);
                af[tm].u[1] = *(const U4*)(p + 16);
            }
#pragma unroll
            for (int tn = 0; tn < 4; ++tn) {
                const _Float16* p = curB + bcol0 + tn * 16 * LDS + kof + kbB;
                bf[tn].u[0] = *(const U4*)(p);
                bf[tn].u[1] = *(const U4*)(p + 16);
            }
#pragma unroll
            for (int tm = 0; tm < 2; ++tm)
#pragma unroll
                for (int tn = 0; tn < 4; ++tn)
                    acc[tm][tn] = __builtin_amdgcn_wmma_f32_16x16x32_f16(
                        false, af[tm].h, false, bf[tn].h,
                        (short)0, acc[tm][tn], false, false);
        }

        if (kn < K_DIM) {
            stage_fence();         // async DMA of next tile complete
            __syncthreads();
            _Float16* t;
            t = curA; curA = nxtA; nxtA = t;
            t = curB; curB = nxtB; nxtB = t;
        }
    }

    // --- fused loss epilogue ---
    // C/D layout: VGPR r, lanes 0-15 -> M=r, N=lane; lanes 16-31 -> M=8+r
    const int mhi = (lane >> 4) * 8;
    float local = 0.f;
#pragma unroll
    for (int tm = 0; tm < 2; ++tm) {
#pragma unroll
        for (int tn = 0; tn < 4; ++tn) {
            const int n = bn0 + wn * 64 + tn * 16 + (lane & 15);
#pragma unroll
            for (int r = 0; r < 8; ++r) {
                const int m = bm0 + wm * 32 + tm * 16 + mhi + r;
                const float d = acc[tm][tn][r] - S[(size_t)m * N_DIM + n];
                local += d * d;
            }
        }
    }
#pragma unroll
    for (int off = 16; off > 0; off >>= 1)
        local += __shfl_xor(local, off, 32);
    if (lane == 0) red[wave] = local;
    __syncthreads();
    if (tid == 0) {
        float s = 0.f;
#pragma unroll
        for (int i = 0; i < 8; ++i) s += red[i];
        atomicAdd(loss_acc, s);
    }
}

// ---------------------------------------------------------------------------
__global__ void zero_kernel(float* p) { p[0] = 0.f; }

__global__ void finalize_kernel(const float* __restrict__ accum,
                                float* __restrict__ out) {
    // loss = sum((pred - int_sum)^2) / (SCALE^2 * M * N)
    out[0] = accum[0] * (1.0f / (255.0f * 255.0f * 2048.0f * 1024.0f));
}

// ---------------------------------------------------------------------------
extern "C" void kernel_launch(void* const* d_in, const int* in_sizes, int n_in,
                              void* d_out, int out_size, void* d_ws, size_t ws_size,
                              hipStream_t stream) {
    const float* latent   = (const float*)d_in[0];  // [2048, 8192]
    const float* true_sum = (const float*)d_in[1];  // [2048, 8192]
    const float* weight   = (const float*)d_in[2];  // [8192, 8192]
    float* out = (float*)d_out;

    // Workspace layout: accum(256B) | A f16 (32MB) | B f16 (16MB) | S f32 (8MB)
    char* ws = (char*)d_ws;
    float*    accum = (float*)ws;
    _Float16* Ah    = (_Float16*)(ws + 256);
    _Float16* Bp    = (_Float16*)(ws + 256 + (size_t)M_DIM * K_DIM * 2);
    float*    S     = (float*)(ws + 256 + (size_t)M_DIM * K_DIM * 2
                                        + (size_t)K_DIM * N_DIM * 2);

    zero_kernel<<<dim3(1), dim3(1), 0, stream>>>(accum);
    pack_weight_kernel<<<dim3(K_DIM / 64, N_DIM / 64), dim3(256), 0, stream>>>(weight, Bp);
    cvt_latent_kernel<<<dim3((M_DIM * K_DIM / 4) / 256), dim3(256), 0, stream>>>(latent, Ah);
    int_sum_kernel<<<dim3((M_DIM * N_DIM) / 256), dim3(256), 0, stream>>>(true_sum, S);
    gemm_loss_kernel<<<dim3(N_DIM / BN, M_DIM / BM), dim3(256), 0, stream>>>(Ah, Bp, S, accum);
    finalize_kernel<<<dim3(1), dim3(1), 0, stream>>>(accum, out);
}
